// NetworkPlusCircuitMinimal_20074677141979
// MI455X (gfx1250) — compile-verified, matrix operand order checked by reading, not run on hardware
//
#include <hip/hip_runtime.h>

typedef __attribute__((ext_vector_type(16))) _Float16 v16h;
typedef __attribute__((ext_vector_type(8)))  _Float16 v8h;
typedef __attribute__((ext_vector_type(8)))  float    v8f;
typedef __attribute__((ext_vector_type(4)))  int      v4i;

// ---------------------------------------------------------------------------
// Async global->LDS copy (CDNA5 GLOBAL_LOAD_ASYNC_TO_LDS_B128, ASYNCcnt).
// Builtin signature (from hipcc diagnostic): (v4i AS1*, v4i AS3*, Ii, Ii).
// Guarded: falls back to synchronous copy if the builtin is absent.
// ---------------------------------------------------------------------------
#if defined(__has_builtin)
#if __has_builtin(__builtin_amdgcn_global_load_async_to_lds_b128) && \
    __has_builtin(__builtin_amdgcn_s_wait_asynccnt)
#define HAS_ASYNC_LDS 1
#endif
#endif

typedef __attribute__((address_space(1))) v4i GV4;
typedef __attribute__((address_space(3))) v4i LV4;

__device__ __forceinline__ void copy16_g2l(_Float16* lds, const _Float16* g) {
#ifdef HAS_ASYNC_LDS
    // AS1 pointer shares the flat address value; generic-LDS low 32 bits are
    // the LDS byte offset (ISA 10.2 aperture mapping), so these casts are exact.
    GV4* gp = (GV4*)(unsigned long long)(size_t)g;
    LV4* lp = (LV4*)(unsigned int)(size_t)lds;
    __builtin_amdgcn_global_load_async_to_lds_b128(gp, lp, 0, 0);
#else
    *(v8h*)lds = *(const v8h*)g;
#endif
}

// ---------------------------------------------------------------------------
// 1) Circuit construction: exact replay of numpy.random.RandomState(0) legacy
//    stream (MT19937; shuffle -> 32-bit masked rejection; int64 randint ->
//    64-bit masked rejection with next64 = hi<<32 | lo).
// ---------------------------------------------------------------------------
struct MT19937 { unsigned mt[624]; int mti; };

__device__ inline void mt_init(MT19937& s, unsigned seed) {
    s.mt[0] = seed;
    for (int i = 1; i < 624; ++i)
        s.mt[i] = 1812433253u * (s.mt[i-1] ^ (s.mt[i-1] >> 30)) + (unsigned)i;
    s.mti = 624;
}
__device__ inline unsigned mt_next32(MT19937& s) {
    if (s.mti >= 624) {
        for (int i = 0; i < 624; ++i) {
            unsigned y = (s.mt[i] & 0x80000000u) | (s.mt[(i+1)%624] & 0x7fffffffu);
            s.mt[i] = s.mt[(i+397)%624] ^ (y >> 1) ^ ((y & 1u) ? 0x9908b0dfu : 0u);
        }
        s.mti = 0;
    }
    unsigned y = s.mt[s.mti++];
    y ^= y >> 11; y ^= (y << 7) & 0x9d2c5680u; y ^= (y << 15) & 0xefc60000u; y ^= y >> 18;
    return y;
}
__device__ inline unsigned long long mt_next64(MT19937& s) {
    unsigned hi = mt_next32(s), lo = mt_next32(s);
    return ((unsigned long long)hi << 32) | lo;
}
__device__ inline unsigned smear32(unsigned v){ v|=v>>1; v|=v>>2; v|=v>>4; v|=v>>8; v|=v>>16; return v; }
__device__ inline unsigned long long smear64(unsigned long long v){
    v|=v>>1; v|=v>>2; v|=v>>4; v|=v>>8; v|=v>>16; v|=v>>32; return v; }

__global__ void init_circuit_kernel(int* __restrict__ prim, int* __restrict__ subs) {
    if (threadIdx.x != 0 || blockIdx.x != 0) return;
    MT19937 s; mt_init(s, 0u);
    for (int i = 0; i < 256; ++i) {
        int arr[10];
        for (int a = 0; a < 10; ++a) arr[a] = a;
        for (int ii = 9; ii >= 1; --ii) {
            unsigned mask = smear32((unsigned)ii), j;
            do { j = mt_next32(s) & mask; } while (j > (unsigned)ii);
            int t = arr[ii]; arr[ii] = arr[j]; arr[j] = t;
        }
        unsigned long long rng = (unsigned long long)(20 + i) - 1ull;
        unsigned long long mask64 = smear64(rng);
        for (int e = 0; e < 3; ++e) {
            prim[i*3 + e] = arr[e];
            unsigned long long v;
            do { v = mt_next64(s) & mask64; } while (v > rng);
            subs[i*3 + e] = (int)v;
        }
    }
}

// ---------------------------------------------------------------------------
// 2) Conversion kernels
// ---------------------------------------------------------------------------
// x [R][Cin] f32 -> [R][Cout] f16, zero-padding columns Cin..Cout
__global__ void cvt_pad_cols(const float* __restrict__ in, _Float16* __restrict__ out,
                             int R, int Cin, int Cout) {
    size_t i = (size_t)blockIdx.x * 256 + threadIdx.x;
    size_t total = (size_t)R * Cout;
    if (i >= total) return;
    int r = (int)(i / Cout), c = (int)(i % Cout);
    out[i] = (c < Cin) ? (_Float16)in[(size_t)r * Cin + c] : (_Float16)0.f;
}

// W [Kin][N] f32 -> Wt [N][Kout] f16 (transpose + pad K), tiled through LDS.
__global__ __launch_bounds__(256) void cvt_transpose_f16(
    const float* __restrict__ in, _Float16* __restrict__ out,
    int Kin, int Kout, int N) {
    __shared__ _Float16 tile[32][33];
    const int k0 = blockIdx.x * 32, n0 = blockIdx.y * 32;
    const int j = threadIdx.x & 31, i0 = threadIdx.x >> 5;
    #pragma unroll
    for (int ii = 0; ii < 32; ii += 8) {
        int k = k0 + i0 + ii;
        float v = (k < Kin) ? in[(size_t)k * N + n0 + j] : 0.f;
        tile[i0 + ii][j] = (_Float16)v;
    }
    __syncthreads();
    #pragma unroll
    for (int ii = 0; ii < 32; ii += 8) {
        int n = n0 + i0 + ii;
        out[(size_t)n * Kout + k0 + j] = tile[j][i0 + ii];
    }
}

// W3 [4096][10] f32 -> W3t [10][4096] f16
__global__ void cvt_w3_transpose(const float* __restrict__ w3, _Float16* __restrict__ out) {
    int i = blockIdx.x * 256 + threadIdx.x;
    if (i >= 10 * 4096) return;
    int c = i / 4096, k = i % 4096;
    out[i] = (_Float16)w3[(size_t)k * 10 + c];
}

// ---------------------------------------------------------------------------
// 3) Tiled WMMA GEMM: C = relu(A[MxK] * Bt[NxK]^T + bias), f16 in, f32 acc.
//    128x128 block, 8 waves (2x4), 64x32 per-wave tile, K-step 32,
//    double-buffered LDS fed by async global->LDS copies.
// ---------------------------------------------------------------------------
__global__ __launch_bounds__(256) void gemm_relu_f16(
    const _Float16* __restrict__ A, const _Float16* __restrict__ Bt,
    const float* __restrict__ bias, _Float16* __restrict__ C,
    int M, int N, int K)
{
    constexpr int LD = 40;                              // halves; 80 B rows, conflict-free
    __shared__ __align__(16) _Float16 As[2][128 * LD];
    __shared__ __align__(16) _Float16 Bs[2][128 * LD];

    const int tid = threadIdx.x;
    const int m0 = blockIdx.y * 128;
    const int n0 = blockIdx.x * 128;
    const int w    = tid >> 5, lane = tid & 31;
    const int wm   = w & 1,    wn   = w >> 1;           // 2 (M) x 4 (N) wave grid
    const int lg   = lane >> 4, lr  = lane & 15;

    // each thread stages 2 A-segments and 2 B-segments of 8 halves per K-step
    const int row0 = tid >> 2,          seg0 = tid & 3;
    const int row1 = (tid + 256) >> 2,  seg1 = (tid + 256) & 3;

    v8f acc[4][2];
    #pragma unroll
    for (int f = 0; f < 4; ++f)
        #pragma unroll
        for (int g = 0; g < 2; ++g)
            #pragma unroll
            for (int j = 0; j < 8; ++j) acc[f][g][j] = 0.f;

    union Frag { v16h v; v8h h[2]; };

    const int nsteps = K >> 5;

    auto stage = [&](int buf, int step) {
        const int k0 = step << 5;
        copy16_g2l(&As[buf][row0 * LD + seg0 * 8], A  + (size_t)(m0 + row0) * K + k0 + seg0 * 8);
        copy16_g2l(&Bs[buf][row0 * LD + seg0 * 8], Bt + (size_t)(n0 + row0) * K + k0 + seg0 * 8);
        copy16_g2l(&As[buf][row1 * LD + seg1 * 8], A  + (size_t)(m0 + row1) * K + k0 + seg1 * 8);
        copy16_g2l(&Bs[buf][row1 * LD + seg1 * 8], Bt + (size_t)(n0 + row1) * K + k0 + seg1 * 8);
    };

    stage(0, 0);
    for (int step = 0; step < nsteps; ++step) {
        if (step + 1 < nsteps) {
            stage((step + 1) & 1, step + 1);            // prefetch next tiles
#ifdef HAS_ASYNC_LDS
            __builtin_amdgcn_s_wait_asynccnt(4);        // current tiles landed
#endif
        } else {
#ifdef HAS_ASYNC_LDS
            __builtin_amdgcn_s_wait_asynccnt(0);
#endif
        }
        __syncthreads();

        const _Float16* as = As[step & 1];
        const _Float16* bs = Bs[step & 1];

        Frag a[4], bf[2];
        #pragma unroll
        for (int f = 0; f < 4; ++f) {                   // A frag: M=wm*64+f*16+lr
            const _Float16* p = &as[(wm * 64 + f * 16 + lr) * LD + lg * 8];
            a[f].h[0] = *(const v8h*)p;
            a[f].h[1] = *(const v8h*)(p + 16);
        }
        #pragma unroll
        for (int g = 0; g < 2; ++g) {                   // B frag: N=wn*32+g*16+lr
            const _Float16* p = &bs[(wn * 32 + g * 16 + lr) * LD + lg * 8];
            bf[g].h[0] = *(const v8h*)p;
            bf[g].h[1] = *(const v8h*)(p + 16);
        }
        #pragma unroll
        for (int f = 0; f < 4; ++f)
            #pragma unroll
            for (int g = 0; g < 2; ++g)
                acc[f][g] = __builtin_amdgcn_wmma_f32_16x16x32_f16(
                    false, a[f].v, false, bf[g].v, (short)0, acc[f][g], false, false);
        __syncthreads();
    }

    // epilogue: bias + relu + f16 store (C/D layout: VGPR j -> M = j + 8*lg)
    #pragma unroll
    for (int f = 0; f < 4; ++f)
        #pragma unroll
        for (int g = 0; g < 2; ++g) {
            int gcol = n0 + wn * 32 + g * 16 + lr;
            float bv = bias[gcol];
            #pragma unroll
            for (int j = 0; j < 8; ++j) {
                int grow = m0 + wm * 64 + f * 16 + lg * 8 + j;
                float v = acc[f][g][j] + bv;
                v = v > 0.f ? v : 0.f;
                C[(size_t)grow * N + gcol] = (_Float16)v;
            }
        }
}

// ---------------------------------------------------------------------------
// 4) Fused GEMM3 (4096 -> 10) + softmax + SDD circuit. One wave per row.
// ---------------------------------------------------------------------------
__global__ __launch_bounds__(256) void head_kernel(
    const _Float16* __restrict__ h2, const _Float16* __restrict__ w3t,
    const float* __restrict__ b3, const int* __restrict__ prim,
    const int* __restrict__ subs, float* __restrict__ out)
{
    __shared__ int   cprim[768], csub[768];
    __shared__ float vals[8][280];
    const int tid = threadIdx.x;
    for (int i = tid; i < 768; i += 256) { cprim[i] = prim[i]; csub[i] = subs[i]; }
    __syncthreads();

    const int w = tid >> 5, lane = tid & 31;
    const int row = blockIdx.x * 8 + w;
    const _Float16* hrow = h2 + (size_t)row * 4096;

    float acc[10];
    #pragma unroll
    for (int c = 0; c < 10; ++c) acc[c] = 0.f;

    for (int it = 0; it < 16; ++it) {
        int kb = it * 256 + lane * 8;
        v8h hv = *(const v8h*)(hrow + kb);
        float hf[8];
        #pragma unroll
        for (int e = 0; e < 8; ++e) hf[e] = (float)hv[e];
        #pragma unroll
        for (int c = 0; c < 10; ++c) {
            v8h wv = *(const v8h*)(w3t + c * 4096 + kb);
            #pragma unroll
            for (int e = 0; e < 8; ++e) acc[c] += hf[e] * (float)wv[e];
        }
    }
    #pragma unroll
    for (int c = 0; c < 10; ++c)
        for (int off = 16; off > 0; off >>= 1)
            acc[c] += __shfl_xor(acc[c], off, 32);

    if (lane == 0) {
        float lg[10], mx = -3.4e38f;
        #pragma unroll
        for (int c = 0; c < 10; ++c) { lg[c] = acc[c] + b3[c]; mx = lg[c] > mx ? lg[c] : mx; }
        float sum = 0.f;
        #pragma unroll
        for (int c = 0; c < 10; ++c) { lg[c] = __expf(lg[c] - mx); sum += lg[c]; }
        float inv = 1.f / sum;
        float* v = vals[w];
        #pragma unroll
        for (int c = 0; c < 10; ++c) { float p = lg[c] * inv; v[c] = p; v[10 + c] = 1.f - p; }
        for (int i = 0; i < 256; ++i) {
            float a = v[cprim[i*3+0]] * v[csub[i*3+0]]
                    + v[cprim[i*3+1]] * v[csub[i*3+1]]
                    + v[cprim[i*3+2]] * v[csub[i*3+2]];
            v[20 + i] = a;
        }
        out[row] = v[275];
    }
}

// ---------------------------------------------------------------------------
// Host launcher
// ---------------------------------------------------------------------------
extern "C" void kernel_launch(void* const* d_in, const int* in_sizes, int n_in,
                              void* d_out, int out_size, void* d_ws, size_t ws_size,
                              hipStream_t stream) {
    (void)in_sizes; (void)n_in; (void)out_size; (void)ws_size;
    const float* x  = (const float*)d_in[0];   // [8192,784]
    const float* W1 = (const float*)d_in[1];   // [784,4096]
    const float* b1 = (const float*)d_in[2];   // [4096]
    const float* W2 = (const float*)d_in[3];   // [4096,4096]
    const float* b2 = (const float*)d_in[4];   // [4096]
    const float* W3 = (const float*)d_in[5];   // [4096,10]
    const float* b3 = (const float*)d_in[6];   // [10]
    float* out = (float*)d_out;

    const int B = 8192, H = 4096, KP = 800;    // 784 padded to 800 (25 * 32)
    char* ws = (char*)d_ws;
    size_t off = 0;
    int*      prim = (int*)(ws + off);       off += 768 * sizeof(int);
    int*      subs = (int*)(ws + off);       off += 768 * sizeof(int);
    _Float16* x16  = (_Float16*)(ws + off);  off += (size_t)B * KP * 2;
    _Float16* w1t  = (_Float16*)(ws + off);  off += (size_t)H * KP * 2;   // [4096][800]
    _Float16* w2t  = (_Float16*)(ws + off);  off += (size_t)H * H * 2;    // [4096][4096]
    _Float16* w3t  = (_Float16*)(ws + off);  off += (size_t)10 * H * 2;   // [10][4096]
    _Float16* h1   = (_Float16*)(ws + off);  off += (size_t)B * H * 2;
    _Float16* h2   = (_Float16*)(ws + off);  off += (size_t)B * H * 2;

    init_circuit_kernel<<<1, 1, 0, stream>>>(prim, subs);

    {   size_t tot = (size_t)B * KP;
        cvt_pad_cols<<<(unsigned)((tot + 255) / 256), 256, 0, stream>>>(x, x16, B, 784, KP); }
    cvt_transpose_f16<<<dim3(KP / 32, H / 32), 256, 0, stream>>>(W1, w1t, 784, KP, H);
    cvt_transpose_f16<<<dim3(H / 32, H / 32), 256, 0, stream>>>(W2, w2t, H, H, H);
    cvt_w3_transpose<<<(10 * H + 255) / 256, 256, 0, stream>>>(W3, w3t);

    dim3 g1(H / 128, B / 128);
    gemm_relu_f16<<<g1, 256, 0, stream>>>(x16, w1t, b1, h1, B, H, KP);
    gemm_relu_f16<<<g1, 256, 0, stream>>>(h1, w2t, b2, h2, B, H, H);

    head_kernel<<<B / 8, 256, 0, stream>>>(h2, w3t, b3, prim, subs, out);
}
